// GNN_5377299054975
// MI455X (gfx1250) — matrix operand order repaired and checked by reading, not compiled, not run on hardware
//
#include <hip/hip_runtime.h>
#include <hip/hip_bf16.h>

// ---------------------------------------------------------------------------
// GNN forward for MI455X (gfx1250, wave32, WMMA).
// All dense layers run through a templated WMMA GEMM using
// v_wmma_f32_16x16x32_f16 (f16 inputs, f32 accumulate). The whole working set
// (h: 4MB, weights ~5MB, edge scratch ~100MB) is L2-resident, so the design
// target is matrix-pipe throughput, not HBM bandwidth.
// ---------------------------------------------------------------------------

#define BATCH   8
#define NNODE   512
#define BN      4096          // BATCH * NNODE
#define DNODE   200
#define HID     256
#define GIN     768           // HID + NNODE
#define STEPS   10

typedef __attribute__((ext_vector_type(16))) _Float16 v16h;
typedef __attribute__((ext_vector_type(8)))  _Float16 v8h;
typedef __attribute__((ext_vector_type(8)))  float    v8f;
typedef __attribute__((ext_vector_type(4)))  float    v4f;

// ---------------------------------------------------------------------------
// Fragment builders (CDNA5 WMMA 16x16x32 f16 layouts, wave32)
//   A 16x32:  lane = 16*g + m.  elements 0..7  = K[k0 + 8g .. +8)
//                               elements 8..15 = K[k0 + 16 + 8g .. +8)
//   B 32x16:  lane = 16*g + n.  elements 0..7 = K[k0 + 8g .. +8), 8..15 = K[k0 + 16 + 8g .. +8) of col n
//   C/D 16x16: lane = 16*g + n, acc element v -> row 8g+v, col n
// ---------------------------------------------------------------------------

__device__ __forceinline__ v16h frag_join(v8h lo, v8h hi) {
  v16h a;
#pragma unroll
  for (int i = 0; i < 8; ++i) { a[i] = lo[i]; a[8 + i] = hi[i]; }
  return a;
}

__device__ __forceinline__ v16h a_frag_f32(const float* p, int k0, int g) {
  const v4f* q0 = (const v4f*)(p + k0 + 8 * g);
  const v4f* q1 = (const v4f*)(p + k0 + 16 + 8 * g);
  v4f x0 = q0[0], x1 = q0[1], y0 = q1[0], y1 = q1[1];
  v16h a;
#pragma unroll
  for (int i = 0; i < 4; ++i) {
    a[i]      = (_Float16)x0[i];
    a[4 + i]  = (_Float16)x1[i];
    a[8 + i]  = (_Float16)y0[i];
    a[12 + i] = (_Float16)y1[i];
  }
  return a;
}

__device__ __forceinline__ v16h a_frag_diff(const float* pr, const float* pc,
                                            int k0, int g) {
  const v4f* r0 = (const v4f*)(pr + k0 + 8 * g);
  const v4f* r1 = (const v4f*)(pr + k0 + 16 + 8 * g);
  const v4f* c0 = (const v4f*)(pc + k0 + 8 * g);
  const v4f* c1 = (const v4f*)(pc + k0 + 16 + 8 * g);
  v4f xr0 = r0[0], xr1 = r0[1], yr0 = r1[0], yr1 = r1[1];
  v4f xc0 = c0[0], xc1 = c0[1], yc0 = c1[0], yc1 = c1[1];
  v16h a;
#pragma unroll
  for (int i = 0; i < 4; ++i) {
    a[i]      = (_Float16)(xr0[i] - xc0[i]);
    a[4 + i]  = (_Float16)(xr1[i] - xc1[i]);
    a[8 + i]  = (_Float16)(yr0[i] - yc0[i]);
    a[12 + i] = (_Float16)(yr1[i] - yc1[i]);
  }
  return a;
}

__device__ __forceinline__ v16h a_frag_f16(const _Float16* p, int k0, int g) {
  v8h lo = *(const v8h*)(p + k0 + 8 * g);
  v8h hi = *(const v8h*)(p + k0 + 16 + 8 * g);
  return frag_join(lo, hi);
}

__device__ __forceinline__ v16h b_frag(const _Float16* w, int ldk, int col,
                                       int k0, int g) {
  const _Float16* p = w + (size_t)col * ldk + k0 + 8 * g;
  v8h lo = *(const v8h*)p;
  v8h hi = *(const v8h*)(p + 16);
  return frag_join(lo, hi);
}

// ---------------------------------------------------------------------------
// Generic WMMA GEMM:  O(M,N) = epi( A(M,K) @ W(N,K)^T + bias )
//   AMODE 0: A = fp32 dense (converted inline)
//   AMODE 1: A = f16 dense scratch
//   AMODE 2: A row e = h[erow[e]] - h[ecol[e]]   (edge gather-diff)
//   EMODE 0: +bias, relu            -> f16 out
//   EMODE 1: +bias                  -> f32 out
//   EMODE 2: +bias + (gW1[col,256+r%512] - gW1[col,256+c]), relu -> f16 out
//   EMODE 3: +bias, sigmoid, multiply f16 out buffer in place (alpha*msg)
//   EMODE 4: +bias                  -> f16 out
// Block = 256 thr (8 waves). Wave w: rows [bx*128+16w, +16), cols [by*64, +64).
// ---------------------------------------------------------------------------
template <int AMODE, int EMODE>
__global__ __launch_bounds__(256) void wmma_gemm(
    const void* Aptr, const float* Hbuf, const int* erow, const int* ecol,
    const _Float16* W, const float* bias, const float* gW1,
    _Float16* outH, float* outF, int M, int K, int N) {
  const int lane = threadIdx.x & 31;
  const int wave = threadIdx.x >> 5;
  const int lrow = lane & 15;
  const int g = lane >> 4;
  const int rowA = blockIdx.x * 128 + wave * 16 + lrow;
  const int rA = rowA < M ? rowA : (M - 1);
  const int tile_n = blockIdx.y * 64;

  const float* pr = nullptr;
  const float* pc = nullptr;
  const _Float16* ph = nullptr;
  if (AMODE == 2) {
    pr = Hbuf + (size_t)erow[rA] * HID;
    pc = Hbuf + (size_t)ecol[rA] * HID;
    __builtin_prefetch(pr, 0, 3);          // gathered rows: WGP-scope prefetch
    __builtin_prefetch(pc, 0, 3);
  } else if (AMODE == 0) {
    pr = (const float*)Aptr + (size_t)rA * K;
  } else {
    ph = (const _Float16*)Aptr + (size_t)rA * K;
  }

  v8f acc0 = {}, acc1 = {}, acc2 = {}, acc3 = {};
  for (int k0 = 0; k0 < K; k0 += 32) {
    v16h a;
    if (AMODE == 2)       a = a_frag_diff(pr, pc, k0, g);
    else if (AMODE == 0)  a = a_frag_f32(pr, k0, g);
    else                  a = a_frag_f16(ph, k0, g);
    v16h b0 = b_frag(W, K, tile_n + 0  + lrow, k0, g);
    v16h b1 = b_frag(W, K, tile_n + 16 + lrow, k0, g);
    v16h b2 = b_frag(W, K, tile_n + 32 + lrow, k0, g);
    v16h b3 = b_frag(W, K, tile_n + 48 + lrow, k0, g);
    acc0 = __builtin_amdgcn_wmma_f32_16x16x32_f16(false, a, false, b0, (short)0, acc0, false, false);
    acc1 = __builtin_amdgcn_wmma_f32_16x16x32_f16(false, a, false, b1, (short)0, acc1, false, false);
    acc2 = __builtin_amdgcn_wmma_f32_16x16x32_f16(false, a, false, b2, (short)0, acc2, false, false);
    acc3 = __builtin_amdgcn_wmma_f32_16x16x32_f16(false, a, false, b3, (short)0, acc3, false, false);
  }

  // ---------------- epilogue ----------------
  v8f accs[4] = {acc0, acc1, acc2, acc3};
  const int col_lane = lane & 15;
  const int row0 = blockIdx.x * 128 + wave * 16 + 8 * g;   // C rows row0..row0+7

  // one-hot gather indices for the g-layer: load once per lane row
  int er8[8], ec8[8];
  if (EMODE == 2) {
#pragma unroll
    for (int v = 0; v < 8; ++v) {
      int orow = row0 + v;
      orow = orow < M ? orow : (M - 1);
      er8[v] = erow[orow] & (NNODE - 1);
      ec8[v] = ecol[orow];
    }
  }

  // per-element epilogue (idx strength-reduced by caller: idx += N per row)
  auto epi = [&](int t, int v, size_t idx, int ocol, float bcol,
                 const float* gRow) {
    float val = accs[t][v] + bcol;
    if (EMODE == 0) {
      val = val > 0.f ? val : 0.f;
      outH[idx] = (_Float16)val;
    } else if (EMODE == 1) {
      outF[idx] = val;
    } else if (EMODE == 2) {
      val += gRow[er8[v]] - gRow[ec8[v]];
      val = val > 0.f ? val : 0.f;
      outH[idx] = (_Float16)val;
    } else if (EMODE == 3) {
      const float s = 1.0f / (1.0f + __expf(-val));
      outH[idx] = (_Float16)((float)outH[idx] * s);
    } else {
      outH[idx] = (_Float16)val;
    }
  };

  if (blockIdx.x * 128 + 128 <= (unsigned)M) {
    // full tile: branchless store stream
#pragma unroll
    for (int t = 0; t < 4; ++t) {
      const int ocol = tile_n + 16 * t + col_lane;
      const float bcol = bias[ocol];
      const float* gRow = (EMODE == 2) ? (gW1 + (size_t)ocol * GIN + HID) : nullptr;
      size_t idx = (size_t)row0 * N + ocol;
#pragma unroll
      for (int v = 0; v < 8; ++v, idx += (size_t)N)
        epi(t, v, idx, ocol, bcol, gRow);
    }
  } else {
#pragma unroll
    for (int t = 0; t < 4; ++t) {
      const int ocol = tile_n + 16 * t + col_lane;
      const float bcol = bias[ocol];
      const float* gRow = (EMODE == 2) ? (gW1 + (size_t)ocol * GIN + HID) : nullptr;
      size_t idx = (size_t)row0 * N + ocol;
#pragma unroll
      for (int v = 0; v < 8; ++v, idx += (size_t)N)
        if (row0 + v < M) epi(t, v, idx, ocol, bcol, gRow);
    }
  }
}

// ----------------------------- support kernels -----------------------------

__global__ void cvt_f16_k(const float* src, _Float16* dst, int n, int cols, int ld) {
  int i = blockIdx.x * 256 + threadIdx.x;
  if (i >= n) return;
  int r = i / cols, c = i - r * cols;
  dst[i] = (_Float16)src[(size_t)r * ld + c];
}

// h init: one-hot @ proj_W^T + proj_b  == column gather of proj_W
__global__ void init_h_k(const int* nodes, const float* projW, const float* projb,
                         float* h) {
  int i = blockIdx.x;      // node row 0..4095
  int f = threadIdx.x;     // 0..255
  int b = i >> 9, j = i & (NNODE - 1);
  float v = projb[f];
  if (j < NNODE - 1) v += projW[(size_t)f * DNODE + nodes[b * (NNODE - 1) + j]];
  h[(size_t)i * HID + f] = v;
}

// CSR row pointers via lower_bound on sorted edges_row
__global__ void rowptr_k(const int* erow, int E, int* rp) {
  int i = blockIdx.x * 256 + threadIdx.x;
  if (i > BN) return;
  int lo = 0, hi = E;
  while (lo < hi) { int mid = (lo + hi) >> 1; if (erow[mid] < i) lo = mid + 1; else hi = mid; }
  rp[i] = lo;
}

__global__ void segsum_k(const _Float16* wmsg, const int* rp, float* merge) {
  int i = blockIdx.x, f = threadIdx.x;
  int e0 = rp[i], e1 = rp[i + 1];
  float s = 0.f;
  for (int e = e0; e < e1; ++e) s += (float)wmsg[(size_t)e * HID + f];
  merge[(size_t)i * HID + f] = s;
}

__global__ void gru_k(const float* gi, const float* gh, float* h) {
  int i = blockIdx.x, f = threadIdx.x;
  size_t b = (size_t)i * GIN;
  float ir = gi[b + f], iz = gi[b + HID + f], in = gi[b + 2 * HID + f];
  float hr = gh[b + f], hz = gh[b + HID + f], hn = gh[b + 2 * HID + f];
  float r = 1.f / (1.f + __expf(-(ir + hr)));
  float z = 1.f / (1.f + __expf(-(iz + hz)));
  float nh = tanhf(in + r * hn);
  size_t idx = (size_t)i * HID + f;
  h[idx] = (1.f - z) * nh + z * h[idx];
}

__global__ void difffin_k(const float* h, float* dfin) {
  int i = blockIdx.x, f = threadIdx.x;
  int last = (i & ~(NNODE - 1)) | (NNODE - 1);
  dfin[(size_t)i * HID + f] = h[(size_t)last * HID + f] - h[(size_t)i * HID + f];
}

// P_A: sigmoid( Z2a . (aW3[0]+aW3[1]) + Z2t . (tW3[0]+tW3[1]) + biases )
__global__ void pa_k(const _Float16* Z2a, const _Float16* Z2t,
                     const float* aW3, const float* ab3,
                     const float* tW3, const float* tb3, float* out) {
  int i = blockIdx.x * 256 + threadIdx.x;
  if (i >= BN) return;
  float acc = ab3[0] + ab3[1] + tb3[0] + tb3[1];
  const _Float16* za = Z2a + (size_t)i * HID;
  const _Float16* zt = Z2t + (size_t)i * HID;
  for (int k = 0; k < HID; ++k)
    acc += (float)za[k] * (aW3[k] + aW3[HID + k]) +
           (float)zt[k] * (tW3[k] + tW3[HID + k]);
  out[i] = 1.f / (1.f + __expf(-acc));
}

// P_X path (M=8, tiny -> plain VALU kernels)
__global__ void px_lin1_k(const float* h, const float* W, const float* b, float* out) {
  int r = blockIdx.x, f = threadIdx.x;
  const float* x = h + ((size_t)r * NNODE + (NNODE - 1)) * HID;
  const float* w = W + (size_t)f * HID;
  float acc = b[f];
  for (int k = 0; k < HID; ++k) acc += x[k] * w[k];
  out[(size_t)r * HID + f] = acc > 0.f ? acc : 0.f;
}
__global__ void px_lin2_k(const float* X, const float* W, const float* b, float* out) {
  int r = blockIdx.x, f = threadIdx.x;
  const float* x = X + (size_t)r * HID;
  const float* w = W + (size_t)f * HID;
  float acc = b[f];
  for (int k = 0; k < HID; ++k) acc += x[k] * w[k];
  out[(size_t)r * HID + f] = acc > 0.f ? acc : 0.f;
}
__global__ void px_lin3_k(const float* X, const float* W, const float* b, float* lg) {
  int r = blockIdx.x, f = threadIdx.x;
  if (f >= DNODE) return;
  const float* x = X + (size_t)r * HID;
  const float* w = W + (size_t)f * HID;
  float acc = b[f];
  for (int k = 0; k < HID; ++k) acc += x[k] * w[k];
  lg[(size_t)r * DNODE + f] = acc;
}
__global__ void px_softmax_k(const float* lg, float* out) {
  int r = threadIdx.x;
  if (r >= BATCH) return;
  const float* l = lg + (size_t)r * DNODE;
  float m = -1e30f;
  for (int k = 0; k < DNODE; ++k) m = fmaxf(m, l[k]);
  float s = 0.f;
  for (int k = 0; k < DNODE; ++k) s += __expf(l[k] - m);
  float inv = 1.f / s;
  for (int k = 0; k < DNODE; ++k) out[(size_t)r * DNODE + k] = __expf(l[k] - m) * inv;
}

// ------------------------------ host wiring --------------------------------

enum {  // setup_inputs dict order; params flattened in insertion order
  IN_NODES = 0, IN_ADJS = 1,
  IN_PROJ_W = 2, IN_PROJ_B = 3,
  IN_A_W1 = 4, IN_A_B1 = 5, IN_A_W2 = 6, IN_A_B2 = 7, IN_A_W3 = 8, IN_A_B3 = 9,
  IN_B_W1 = 10, IN_B_B1 = 11, IN_B_W2 = 12, IN_B_B2 = 13, IN_B_W3 = 14, IN_B_B3 = 15,
  IN_T_W1 = 16, IN_T_B1 = 17, IN_T_W2 = 18, IN_T_B2 = 19, IN_T_W3 = 20, IN_T_B3 = 21,
  IN_F_W1 = 22, IN_F_B1 = 23, IN_F_W2 = 24, IN_F_B2 = 25,
  IN_G_W1 = 26, IN_G_B1 = 27, IN_G_W2 = 28, IN_G_B2 = 29,
  IN_GRU_WIH = 30, IN_GRU_WHH = 31, IN_GRU_BIH = 32, IN_GRU_BHH = 33,
  IN_EROW = 34, IN_ECOL = 35,
};

extern "C" void kernel_launch(void* const* d_in, const int* in_sizes, int n_in,
                              void* d_out, int out_size, void* d_ws, size_t ws_size,
                              hipStream_t stream) {
  const int* nodes = (const int*)d_in[IN_NODES];
  const int* erow = (const int*)d_in[IN_EROW];
  const int* ecol = (const int*)d_in[IN_ECOL];
  const size_t E = (size_t)in_sizes[IN_EROW];
  const size_t Epad = E < 2 * BN ? (size_t)(2 * BN) : E;  // reuse-region floor

  const float* projW = (const float*)d_in[IN_PROJ_W];
  const float* projB = (const float*)d_in[IN_PROJ_B];
  const float* gW1f  = (const float*)d_in[IN_G_W1];

  // ---- workspace carve ----
  char* ws = (char*)d_ws;
  size_t off = 0;
  auto alloc = [&](size_t bytes) -> void* {
    void* p = ws + off;
    off = (off + bytes + 255) & ~(size_t)255;
    return p;
  };
  float*     h      = (float*)alloc((size_t)BN * HID * 4);
  _Float16*  fW1h   = (_Float16*)alloc((size_t)HID * HID * 2);
  _Float16*  fW2h   = (_Float16*)alloc((size_t)HID * HID * 2);
  _Float16*  gW1ph  = (_Float16*)alloc((size_t)GIN * HID * 2);  // g_W1[:, :256]
  _Float16*  gW2h   = (_Float16*)alloc((size_t)HID * GIN * 2);
  _Float16*  wihh   = (_Float16*)alloc((size_t)GIN * HID * 2);
  _Float16*  whhh   = (_Float16*)alloc((size_t)GIN * HID * 2);
  _Float16*  aW1h   = (_Float16*)alloc((size_t)HID * HID * 2);
  _Float16*  aW2h   = (_Float16*)alloc((size_t)HID * HID * 2);
  _Float16*  tW1h   = (_Float16*)alloc((size_t)HID * HID * 2);
  _Float16*  tW2h   = (_Float16*)alloc((size_t)HID * HID * 2);
  _Float16*  X1     = (_Float16*)alloc(Epad * HID * 2);      // also Za | Zt
  _Float16*  Y1     = (_Float16*)alloc(Epad * GIN * 2);      // also Z2a | Z2t
  _Float16*  msgs   = (_Float16*)alloc(E * HID * 2);         // msgs -> alpha*msgs
  float*     gi     = (float*)alloc((size_t)BN * GIN * 4);   // also diff_fin
  float*     gh     = (float*)alloc((size_t)BN * GIN * 4);
  float*     merge  = (float*)alloc((size_t)BN * HID * 4);   // also P_X scratch
  int*       rp     = (int*)alloc((size_t)(BN + 1) * 4);

  const dim3 blk(256);
  auto cvt = [&](const float* src, _Float16* dst, int rows, int cols, int ld) {
    int n = rows * cols;
    cvt_f16_k<<<dim3((n + 255) / 256), blk, 0, stream>>>(src, dst, n, cols, ld);
  };

  // ---- one-time weight conversion (fp32 -> f16, K-contiguous rows) ----
  cvt((const float*)d_in[IN_F_W1], fW1h, HID, HID, HID);
  cvt((const float*)d_in[IN_F_W2], fW2h, HID, HID, HID);
  cvt(gW1f,                        gW1ph, GIN, HID, GIN);    // first 256 cols
  cvt((const float*)d_in[IN_G_W2], gW2h, HID, GIN, GIN);
  cvt((const float*)d_in[IN_GRU_WIH], wihh, GIN, HID, HID);
  cvt((const float*)d_in[IN_GRU_WHH], whhh, GIN, HID, HID);
  cvt((const float*)d_in[IN_A_W1], aW1h, HID, HID, HID);
  cvt((const float*)d_in[IN_A_W2], aW2h, HID, HID, HID);
  cvt((const float*)d_in[IN_T_W1], tW1h, HID, HID, HID);
  cvt((const float*)d_in[IN_T_W2], tW2h, HID, HID, HID);

  // ---- h0 projection (one-hot => gather) + CSR rowptr ----
  init_h_k<<<dim3(BN), blk, 0, stream>>>(nodes, projW, projB, h);
  rowptr_k<<<dim3((BN + 1 + 255) / 256), blk, 0, stream>>>(erow, (int)E, rp);

  const unsigned gEx = (unsigned)((E + 127) / 128);
  const dim3 gE256(gEx, HID / 64), gE768(gEx, GIN / 64);
  const dim3 gN256(BN / 128, HID / 64), gN768(BN / 128, GIN / 64);

  // ---- message-passing steps ----
  for (int s = 0; s < STEPS; ++s) {
    // X1 = relu(diff @ f_W1^T + f_b1)         (E,256) f16
    wmma_gemm<2, 0><<<gE256, blk, 0, stream>>>(
        nullptr, h, erow, ecol, fW1h, (const float*)d_in[IN_F_B1],
        nullptr, X1, nullptr, (int)E, HID, HID);
    // msgs = X1 @ f_W2^T + f_b2               (E,256) f16
    wmma_gemm<1, 4><<<gE256, blk, 0, stream>>>(
        X1, nullptr, nullptr, nullptr, fW2h, (const float*)d_in[IN_F_B2],
        nullptr, msgs, nullptr, (int)E, HID, HID);
    // Y1 = relu(diff @ gW1[:, :256]^T + onehot-gather + g_b1)   (E,768) f16
    wmma_gemm<2, 2><<<gE768, blk, 0, stream>>>(
        nullptr, h, erow, ecol, gW1ph, (const float*)d_in[IN_G_B1],
        gW1f, Y1, nullptr, (int)E, HID, GIN);
    // msgs *= sigmoid(Y1 @ g_W2^T + g_b2)     in place
    wmma_gemm<1, 3><<<gE256, blk, 0, stream>>>(
        Y1, nullptr, nullptr, nullptr, gW2h, (const float*)d_in[IN_G_B2],
        nullptr, msgs, nullptr, (int)E, GIN, HID);
    // merge = segment_sum(alpha*msgs)
    segsum_k<<<dim3(BN), blk, 0, stream>>>(msgs, rp, merge);
    // gi = merge @ Wih^T + bih ; gh = h @ Whh^T + bhh
    wmma_gemm<0, 1><<<gN768, blk, 0, stream>>>(
        merge, nullptr, nullptr, nullptr, wihh, (const float*)d_in[IN_GRU_BIH],
        nullptr, nullptr, gi, BN, HID, GIN);
    wmma_gemm<0, 1><<<gN768, blk, 0, stream>>>(
        h, nullptr, nullptr, nullptr, whhh, (const float*)d_in[IN_GRU_BHH],
        nullptr, nullptr, gh, BN, HID, GIN);
    gru_k<<<dim3(BN), blk, 0, stream>>>(gi, gh, h);
  }

  // ---- outputs ----
  float* out = (float*)d_out;           // [0,4096) = P_A, [4096,5696) = P_X
  float* dfin = gi;                     // reuse
  _Float16* Za  = X1;
  _Float16* Zt  = X1 + (size_t)BN * HID;
  _Float16* Z2a = Y1;
  _Float16* Z2t = Y1 + (size_t)BN * HID;

  difffin_k<<<dim3(BN), blk, 0, stream>>>(h, dfin);
  wmma_gemm<0, 0><<<gN256, blk, 0, stream>>>(
      dfin, nullptr, nullptr, nullptr, aW1h, (const float*)d_in[IN_A_B1],
      nullptr, Za, nullptr, BN, HID, HID);
  wmma_gemm<1, 0><<<gN256, blk, 0, stream>>>(
      Za, nullptr, nullptr, nullptr, aW2h, (const float*)d_in[IN_A_B2],
      nullptr, Z2a, nullptr, BN, HID, HID);
  wmma_gemm<0, 0><<<gN256, blk, 0, stream>>>(
      dfin, nullptr, nullptr, nullptr, tW1h, (const float*)d_in[IN_T_B1],
      nullptr, Zt, nullptr, BN, HID, HID);
  wmma_gemm<1, 0><<<gN256, blk, 0, stream>>>(
      Zt, nullptr, nullptr, nullptr, tW2h, (const float*)d_in[IN_T_B2],
      nullptr, Z2t, nullptr, BN, HID, HID);
  pa_k<<<dim3(BN / 256), blk, 0, stream>>>(
      Z2a, Z2t, (const float*)d_in[IN_A_W3], (const float*)d_in[IN_A_B3],
      (const float*)d_in[IN_T_W3], (const float*)d_in[IN_T_B3], out);

  float* pxX1 = merge;                  // reuse merge as P_X scratch
  float* pxX2 = merge + BATCH * HID;
  float* pxLg = merge + 2 * BATCH * HID;
  px_lin1_k<<<dim3(BATCH), blk, 0, stream>>>(
      h, (const float*)d_in[IN_B_W1], (const float*)d_in[IN_B_B1], pxX1);
  px_lin2_k<<<dim3(BATCH), blk, 0, stream>>>(
      pxX1, (const float*)d_in[IN_B_W2], (const float*)d_in[IN_B_B2], pxX2);
  px_lin3_k<<<dim3(BATCH), blk, 0, stream>>>(
      pxX2, (const float*)d_in[IN_B_W3], (const float*)d_in[IN_B_B3], pxLg);
  px_softmax_k<<<dim3(1), dim3(32), 0, stream>>>(pxLg, out + BN);

  (void)n_in; (void)out_size; (void)ws_size; (void)d_in;
}
